// AttentionGated_65197603554154
// MI455X (gfx1250) — matrix operand order repaired and checked by reading, not compiled, not run on hardware
//
#include <hip/hip_runtime.h>
#include <hip/hip_bf16.h>

typedef __attribute__((ext_vector_type(16))) _Float16 v16h;
typedef __attribute__((ext_vector_type(8)))  _Float16 v8h;
typedef __attribute__((ext_vector_type(4)))  _Float16 v4h;
typedef __attribute__((ext_vector_type(8)))  float    v8f;

#define NROWS     50000
#define DIM       768
#define DATT      128
#define TILE_ROWS 32
#define PADK      776   // 768 + 8 halves padding to break LDS bank alignment

union V16H { v16h v; v8h h2[2]; _Float16 e[16]; };
union V8F  { v8f  v; float e[8]; };
union V4H  { v4h  v; _Float16 e[4]; };
union U32H { uint32_t u; _Float16 e[2]; };

// ---------------------------------------------------------------------------
// fp32 -> f16 weight conversion (runs once per launch, tiny)
// ---------------------------------------------------------------------------
__global__ void cvt_f32_f16_kernel(const float* __restrict__ src,
                                   _Float16* __restrict__ dst, int n) {
  int i = blockIdx.x * blockDim.x + threadIdx.x;
  if (i < n) dst[i] = (_Float16)src[i];
}

// ---------------------------------------------------------------------------
// Emit one 16x16 D tile: bias + relu, write f16 into LDS h tile.
// D layout: VGPR i -> (M=i, N=lane) lanes 0-15 ; (M=i+8, N=lane-16) lanes 16-31
// ---------------------------------------------------------------------------
__device__ __forceinline__ void emit_h_tile(const v8f& c, _Float16* hsh,
                                            const float* __restrict__ bfeat,
                                            int ecol, int rbase, int hi) {
  float bias = bfeat[ecol];
  V8F uc; uc.v = c;
#pragma unroll
  for (int i = 0; i < 8; ++i) {
    float v = uc.e[i] + bias;
    v = fmaxf(v, 0.f);
    hsh[(rbase + i + hi * 8) * PADK + ecol] = (_Float16)v;
  }
}

// ---------------------------------------------------------------------------
// Fused: h = relu(x Wf^T + b)  (WMMA, f16 in / f32 acc)
//        a = relu(h Wa^T), b = sigmoid(h Wb^T), logits += (a*b) . Wc
// One block = 32 rows; 8 wave32. x staged once through LDS (f16) so the
// WMMA hot loop is branch-free: ds_load_b128 A + global b128 B + wmma.
// ---------------------------------------------------------------------------
__global__ __launch_bounds__(256) void fused_gated_attn_kernel(
    const float*    __restrict__ x,
    const _Float16* __restrict__ Wf,
    const float*    __restrict__ bfeat,
    const _Float16* __restrict__ Wa,
    const _Float16* __restrict__ Wb,
    const float*    __restrict__ Wc,
    _Float16*       __restrict__ hout,
    float*          __restrict__ logits) {
  extern __shared__ _Float16 smem[];
  _Float16* xsh = smem;                          // [TILE_ROWS][PADK]
  _Float16* hsh = smem + TILE_ROWS * PADK;       // [TILE_ROWS][PADK]
  const int tid  = threadIdx.x;
  const int wave = tid >> 5;
  const int lane = tid & 31;
  const int l16  = lane & 15;
  const int hi   = lane >> 4;  // 0 or 1 (half-wave)
  const int rowBase = blockIdx.x * TILE_ROWS;

  // ---------------- Stage 0: x tile -> LDS f16 (coalesced, OOB = 0) -------
  {
    const int QD = DIM / 4;  // 192 float4 per row
    for (int idx = tid; idx < TILE_ROWS * QD; idx += 256) {
      int r = idx / QD;
      int q = idx - r * QD;
      int grow = rowBase + r;
      V4H h4;
      if (grow < NROWS) {
        float4 f = reinterpret_cast<const float4*>(x + (size_t)grow * DIM)[q];
        h4.e[0] = (_Float16)f.x; h4.e[1] = (_Float16)f.y;
        h4.e[2] = (_Float16)f.z; h4.e[3] = (_Float16)f.w;
      } else {
        h4.e[0] = h4.e[1] = h4.e[2] = h4.e[3] = (_Float16)0.f;
      }
      *reinterpret_cast<v4h*>(xsh + r * PADK + q * 4) = h4.v;
    }
  }
  __syncthreads();

  // ---------------- Stage 1: h = relu(x Wf^T + b) -----------------
  {
    const int rs = wave & 1;   // row strip 0..1 (16 rows each)
    const int eq = wave >> 1;  // output-feature quarter 0..3 (192 cols each)
    const _Float16* arow = xsh + (rs * 16 + l16) * PADK;

    for (int grp = 0; grp < 3; ++grp) {
      const int e0 = eq * 192 + grp * 64;  // 4 e-tiles of 16 per group
      v8f c0 = {}; v8f c1 = {}; v8f c2 = {}; v8f c3 = {};
#pragma unroll 4
      for (int k0 = 0; k0 < DIM; k0 += 32) {
        // A fragment (16x32 f16): lane<16 row=l16 K{k0..k0+15};
        // lane>=16 same rows, K shifted by +16.  Two ds_load_b128.
        const int kb = k0 + hi * 16;
        V16H ua;
        ua.h2[0] = *reinterpret_cast<const v8h*>(arow + kb);
        ua.h2[1] = *reinterpret_cast<const v8h*>(arow + kb + 8);
        // B fragments (32x16 f16): lane col = e0+t*16+l16, K contiguous from
        // k0 + hi*16 -> one 32B row-major load per tile from L2-resident Wf.
        const _Float16* wp = Wf + (size_t)(e0 + l16) * DIM + k0 + hi * 16;
        v16h b0 = *reinterpret_cast<const v16h*>(wp);
        v16h b1 = *reinterpret_cast<const v16h*>(wp + 16 * DIM);
        v16h b2 = *reinterpret_cast<const v16h*>(wp + 32 * DIM);
        v16h b3 = *reinterpret_cast<const v16h*>(wp + 48 * DIM);
        c0 = __builtin_amdgcn_wmma_f32_16x16x32_f16(false, ua.v, false, b0, (short)0, c0, false, false);
        c1 = __builtin_amdgcn_wmma_f32_16x16x32_f16(false, ua.v, false, b1, (short)0, c1, false, false);
        c2 = __builtin_amdgcn_wmma_f32_16x16x32_f16(false, ua.v, false, b2, (short)0, c2, false, false);
        c3 = __builtin_amdgcn_wmma_f32_16x16x32_f16(false, ua.v, false, b3, (short)0, c3, false, false);
      }
      emit_h_tile(c0, hsh, bfeat, e0 +  0 + l16, rs * 16, hi);
      emit_h_tile(c1, hsh, bfeat, e0 + 16 + l16, rs * 16, hi);
      emit_h_tile(c2, hsh, bfeat, e0 + 32 + l16, rs * 16, hi);
      emit_h_tile(c3, hsh, bfeat, e0 + 48 + l16, rs * 16, hi);
    }
  }
  __syncthreads();

  // ---------------- Coalesced h -> global (f16 row-major) -----------------
  {
    const uint32_t* s = reinterpret_cast<const uint32_t*>(hsh);
    uint32_t* g = reinterpret_cast<uint32_t*>(hout);
    for (int idx = tid; idx < TILE_ROWS * (DIM / 2); idx += 256) {
      int r  = idx / (DIM / 2);
      int cd = idx - r * (DIM / 2);
      int grow = rowBase + r;
      if (grow < NROWS)
        g[(size_t)grow * (DIM / 2) + cd] = s[r * (PADK / 2) + cd];
    }
  }

  // ---------------- Stage 2: a, b, logits -----------------
  {
    const int rs2 = wave & 1;  // row strip 0..1
#pragma unroll
    for (int j = 0; j < 2; ++j) {
      const int t = (wave >> 1) * 2 + j;  // katt tile 0..7 (128/16)
      v8f ca = {}; v8f cb = {};
      const _Float16* arow = hsh + (rs2 * 16 + l16) * PADK;
      const _Float16* wpa  = Wa + (size_t)(t * 16 + l16) * DIM + hi * 16;
      const _Float16* wpb  = Wb + (size_t)(t * 16 + l16) * DIM + hi * 16;
#pragma unroll 4
      for (int k0 = 0; k0 < DIM; k0 += 32) {
        const int kb = k0 + hi * 16;
        V16H ua;
        ua.h2[0] = *reinterpret_cast<const v8h*>(arow + kb);       // ds_load_b128
        ua.h2[1] = *reinterpret_cast<const v8h*>(arow + kb + 8);   // ds_load_b128
        v16h ba = *reinterpret_cast<const v16h*>(wpa + k0);
        v16h bb = *reinterpret_cast<const v16h*>(wpb + k0);
        ca = __builtin_amdgcn_wmma_f32_16x16x32_f16(false, ua.v, false, ba, (short)0, ca, false, false);
        cb = __builtin_amdgcn_wmma_f32_16x16x32_f16(false, ua.v, false, bb, (short)0, cb, false, false);
      }
      float wc = Wc[t * 16 + l16];
      V8F uca, ucb; uca.v = ca; ucb.v = cb;
#pragma unroll
      for (int i = 0; i < 8; ++i) {
        float av = fmaxf(uca.e[i], 0.f);
        float bv = 1.f / (1.f + __expf(-ucb.e[i]));
        float p = av * bv * wc;
        // reduce across the 16 lanes that hold one row's katt columns
        p += __shfl_xor(p, 8, 16);
        p += __shfl_xor(p, 4, 16);
        p += __shfl_xor(p, 2, 16);
        p += __shfl_xor(p, 1, 16);
        if (l16 == 0) {
          int grow = rowBase + rs2 * 16 + i + hi * 8;
          if (grow < NROWS) atomicAdd(&logits[grow], p);
        }
      }
    }
  }
}

// ---------------------------------------------------------------------------
// Softmax stats: stats[0] = max(logits), stats[1] = 1/sum(exp(logit - max))
// 200 KB of L2-resident data -> single block is plenty.
// ---------------------------------------------------------------------------
__global__ __launch_bounds__(256) void softmax_stats_kernel(
    const float* __restrict__ logits, float* __restrict__ stats, int n) {
  __shared__ float red[256];
  const int tid = threadIdx.x;
  float m = -3.402823466e+38f;
  for (int i = tid; i < n; i += 256) m = fmaxf(m, logits[i]);
  red[tid] = m; __syncthreads();
  for (int s = 128; s > 0; s >>= 1) {
    if (tid < s) red[tid] = fmaxf(red[tid], red[tid + s]);
    __syncthreads();
  }
  float gmax = red[0]; __syncthreads();
  float sum = 0.f;
  for (int i = tid; i < n; i += 256) sum += __expf(logits[i] - gmax);
  red[tid] = sum; __syncthreads();
  for (int s = 128; s > 0; s >>= 1) {
    if (tid < s) red[tid] += red[tid + s];
    __syncthreads();
  }
  if (tid == 0) { stats[0] = gmax; stats[1] = 1.f / red[0]; }
}

// ---------------------------------------------------------------------------
// out[e] = sum_n softmax(logit)[n] * h[n][e]   (h is f16; out pre-zeroed)
// 384 threads, each owns one dword (2 cols) -> full-row coalesced b32 loads.
// ---------------------------------------------------------------------------
__global__ __launch_bounds__(384) void weighted_sum_kernel(
    const _Float16* __restrict__ h, const float* __restrict__ logits,
    const float* __restrict__ stats, float* __restrict__ out, int n) {
  const int tid = threadIdx.x;  // dword column 0..383
  const int rows_per = (n + (int)gridDim.x - 1) / (int)gridDim.x;
  const int r0 = blockIdx.x * rows_per;
  const int r1 = (r0 + rows_per < n) ? (r0 + rows_per) : n;
  const float gmax = stats[0], inv = stats[1];
  float a0 = 0.f, a1 = 0.f;
  for (int r = r0; r < r1; ++r) {
    float w = __expf(logits[r] - gmax) * inv;
    U32H d;
    d.u = reinterpret_cast<const uint32_t*>(h + (size_t)r * DIM)[tid];
    a0 += w * (float)d.e[0];
    a1 += w * (float)d.e[1];
  }
  atomicAdd(&out[2 * tid],     a0);
  atomicAdd(&out[2 * tid + 1], a1);
}

// ---------------------------------------------------------------------------
extern "C" void kernel_launch(void* const* d_in, const int* in_sizes, int n_in,
                              void* d_out, int out_size, void* d_ws, size_t ws_size,
                              hipStream_t stream) {
  (void)in_sizes; (void)n_in; (void)out_size; (void)ws_size;
  const float* x  = (const float*)d_in[0];
  const float* Wf = (const float*)d_in[1];
  const float* bf = (const float*)d_in[2];
  const float* Wa = (const float*)d_in[3];
  const float* Wb = (const float*)d_in[4];
  const float* Wc = (const float*)d_in[5];
  float* out = (float*)d_out;

  // workspace carve-out (256B aligned)
  char* ws = (char*)d_ws;
  size_t off = 0;
  auto carve = [&](size_t bytes) {
    char* p = ws + off;
    off = (off + bytes + 255) & ~(size_t)255;
    return p;
  };
  _Float16* Wf_h  = (_Float16*)carve((size_t)DIM * DIM * 2);
  _Float16* Wa_h  = (_Float16*)carve((size_t)DATT * DIM * 2);
  _Float16* Wb_h  = (_Float16*)carve((size_t)DATT * DIM * 2);
  float*    logit = (float*)   carve((size_t)NROWS * 4);
  float*    stats = (float*)   carve(256);
  _Float16* hbuf  = (_Float16*)carve((size_t)NROWS * DIM * 2);

  cvt_f32_f16_kernel<<<(DIM * DIM + 255) / 256, 256, 0, stream>>>(Wf, Wf_h, DIM * DIM);
  cvt_f32_f16_kernel<<<(DATT * DIM + 255) / 256, 256, 0, stream>>>(Wa, Wa_h, DATT * DIM);
  cvt_f32_f16_kernel<<<(DATT * DIM + 255) / 256, 256, 0, stream>>>(Wb, Wb_h, DATT * DIM);
  hipMemsetAsync(logit, 0, (size_t)NROWS * sizeof(float), stream);
  hipMemsetAsync(out, 0, (size_t)DIM * sizeof(float), stream);

  const int nblocks = (NROWS + TILE_ROWS - 1) / TILE_ROWS;  // 1563
  const size_t lds = (size_t)2 * TILE_ROWS * PADK * sizeof(_Float16);  // ~97 KB
  fused_gated_attn_kernel<<<nblocks, 256, lds, stream>>>(
      x, Wf_h, bf, Wa_h, Wb_h, Wc, hbuf, logit);
  softmax_stats_kernel<<<1, 256, 0, stream>>>(logit, stats, NROWS);
  weighted_sum_kernel<<<200, 384, 0, stream>>>(hbuf, logit, stats, out, NROWS);
}
